// EfficientViTBlock_2087354106483
// MI455X (gfx1250) — matrix operand order repaired and checked by reading, not compiled, not run on hardware
//
#include <hip/hip_runtime.h>
#include <hip/hip_bf16.h>

typedef unsigned short u16;
typedef __attribute__((ext_vector_type(16))) __bf16 v16bf;
typedef __attribute__((ext_vector_type(8)))  float  v8f;
typedef __attribute__((ext_vector_type(4)))  unsigned int v4u;

union Frag {
  v16bf v;
  u16   u[16];
  v4u   q[2];
};

__device__ __forceinline__ u16 f2bf(float f) {
  unsigned int x = __float_as_uint(f);
  x += 0x7fffu + ((x >> 16) & 1u);           // round-to-nearest-even
  return (u16)(x >> 16);
}
__device__ __forceinline__ float bf2f(u16 h) {
  return __uint_as_float(((unsigned int)h) << 16);
}

// ---------------------------------------------------------------- conversions
__global__ __launch_bounds__(256) void k_cvt(const float* __restrict__ in,
                                             u16* __restrict__ out, int n) {
  int i = blockIdx.x * 256 + threadIdx.x;
  if (i < n) out[i] = f2bf(in[i]);
}

// x (B,C=256,HW=4096) f32  ->  xT (B*HW, 256) bf16  via 64x64 LDS tile transpose
__global__ __launch_bounds__(256) void k_cvtT(const float* __restrict__ in,
                                              u16* __restrict__ outT) {
  __shared__ u16 t[64][65];
  int n0 = blockIdx.x * 64, c0 = blockIdx.y * 64, b = blockIdx.z;
  int tid = threadIdx.x;
#pragma unroll
  for (int j = 0; j < 16; ++j) {
    int lin = j * 256 + tid;
    int r = lin >> 6, col = lin & 63;
    t[r][col] = f2bf(in[((size_t)(b * 256 + c0 + r)) * 4096 + n0 + col]);
  }
  __syncthreads();
#pragma unroll
  for (int j = 0; j < 16; ++j) {
    int lin = j * 256 + tid;
    int r = lin >> 6, col = lin & 63;
    outT[((size_t)b * 4096 + n0 + r) * 256 + c0 + col] = t[col][r];
  }
}

__global__ __launch_bounds__(256) void k_zero(float* __restrict__ p, int n) {
  int i = blockIdx.x * 256 + threadIdx.x;
  if (i < n) p[i] = 0.f;
}

// ---------------------------------------------------------------- WMMA GEMM
// out[b, m, n] = sum_k A[m,k] * XT[b*HW+n, k]
//   A : MxK bf16 row-major,   XT : [B*HW][K] bf16 (activation, n-major)
// MODE 0: plain -> outF (f32, [b][m][n])
// MODE 1: + p0[m] bias, SiLU -> outH (bf16, [n][m])
// MODE 2: BN(p0..p3) + resid([b][m][n]) -> outF f32 [b][m][n] and outH bf16 [n][m]
// MODE 3: + p0[m] bias + resid([b][m][n]) -> outF f32 [b][m][n]
template<int MODE>
__global__ __launch_bounds__(256) void k_gemm(
    const u16* __restrict__ A, const u16* __restrict__ XT,
    int M, int K, int HW,
    float* __restrict__ outF, u16* __restrict__ outH,
    const float* __restrict__ p0, const float* __restrict__ p1,
    const float* __restrict__ p2, const float* __restrict__ p3,
    const float* __restrict__ resid)
{
  constexpr int BM = 128, BN = 128, BK = 32, STR = BK + 8;
  __shared__ __align__(16) u16 ldsX[2][BN * STR];

  const int tid  = threadIdx.x;
  const int lane = tid & 31;
  const int wid  = tid >> 5;
  const int wm   = wid >> 1;           // 0..3  (32 rows each)
  const int wn   = wid & 1;            // 0..1  (64 cols each)
  const int half = lane >> 4;
  const int l15  = lane & 15;

  const int nBlk = blockIdx.x * BN;    // flat n over B*HW
  const int b    = nBlk / HW;
  const int nInB = nBlk - b * HW;
  const int mBlk = blockIdx.y * BM;
  const u16* Xrow = XT + (size_t)nBlk * K;   // 128 rows, stride K

  v8f acc[2][4];
#pragma unroll
  for (int i = 0; i < 2; ++i)
#pragma unroll
    for (int j = 0; j < 4; ++j) acc[i][j] = (v8f){0.f,0.f,0.f,0.f,0.f,0.f,0.f,0.f};

  const int srow = tid >> 1;           // 0..127 : n row of tile
  const int sseg = (tid & 1) << 4;     // 0 or 16 elements within row

  // async global->LDS stage of one 128x32 bf16 tile (32 B per thread)
  auto stage = [&](int k0s, int bufi) {
    const u16* src = Xrow + (size_t)srow * K + k0s + sseg;
    unsigned lofs = (unsigned)(unsigned long long)&ldsX[bufi][srow * STR + sseg];
    unsigned long long ga = (unsigned long long)src;
    asm volatile(
        "global_load_async_to_lds_b128 %0, %1, off\n\t"
        "global_load_async_to_lds_b128 %0, %1, off offset:16"
        :: "v"(lofs), "v"(ga) : "memory");
  };

  stage(0, 0);
  asm volatile("s_wait_asynccnt 0x0" ::: "memory");
  __syncthreads();

  int buf = 0;
  for (int k0 = 0; k0 < K; k0 += BK) {
    if (k0 + BK < K) stage(k0 + BK, buf ^ 1);

    // A fragments (16x32 bf16 layout): lane<16 -> K 0..7 & 16..23, lane>=16 -> K 8..15 & 24..31
    Frag a0, a1;
    {
      const u16* ap = A + (size_t)(mBlk + wm * 32 + l15) * K + k0 + half * 8;
      a0.q[0] = *(const v4u*)ap;
      a0.q[1] = *(const v4u*)(ap + 16);
      const u16* ap1 = ap + (size_t)16 * K;
      a1.q[0] = *(const v4u*)ap1;
      a1.q[1] = *(const v4u*)(ap1 + 16);
    }

    // B fragments (32x16): lane<16 holds K 0..7 & 16..23 of column n, lane>=16 holds K 8..15 & 24..31
#pragma unroll
    for (int nf = 0; nf < 4; ++nf) {
      Frag bf;
      const u16* lp = &ldsX[buf][(wn * 64 + nf * 16 + l15) * STR + half * 8];
      bf.q[0] = *(const v4u*)lp;
      bf.q[1] = *(const v4u*)(lp + 16);
      acc[0][nf] = __builtin_amdgcn_wmma_f32_16x16x32_bf16(
          false, a0.v, false, bf.v, (short)0, acc[0][nf], false, false);
      acc[1][nf] = __builtin_amdgcn_wmma_f32_16x16x32_bf16(
          false, a1.v, false, bf.v, (short)0, acc[1][nf], false, false);
    }

    asm volatile("s_wait_asynccnt 0x0" ::: "memory");
    __syncthreads();
    buf ^= 1;
  }

  // epilogue: C layout VGPR r, lanes0-15 -> M=r, lanes16-31 -> M=r+8; N = lane&15
#pragma unroll
  for (int mf = 0; mf < 2; ++mf) {
#pragma unroll
    for (int nf = 0; nf < 4; ++nf) {
      int mRow = mBlk + wm * 32 + mf * 16 + half * 8;
      int nLoc = nInB + wn * 64 + nf * 16 + l15;
      size_t ng = (size_t)b * HW + nLoc;
      if (MODE == 0) {
#pragma unroll
        for (int r = 0; r < 8; ++r)
          outF[((size_t)b * M + mRow + r) * HW + nLoc] = acc[mf][nf][r];
      } else if (MODE == 1) {
        union { u16 u[8]; v4u q; } pk;
#pragma unroll
        for (int r = 0; r < 8; ++r) {
          float v = acc[mf][nf][r] + p0[mRow + r];
          v = v / (1.f + __expf(-v));                    // SiLU
          pk.u[r] = f2bf(v);
        }
        *(v4u*)(outH + ng * M + mRow) = pk.q;
      } else if (MODE == 2) {
        union { u16 u[8]; v4u q; } pk;
#pragma unroll
        for (int r = 0; r < 8; ++r) {
          int m = mRow + r;
          float inv_std = rsqrtf(p3[m] + 1e-5f);
          float v = (acc[mf][nf][r] - p2[m]) * (p0[m] * inv_std) + p1[m];
          float y = resid[((size_t)b * M + m) * HW + nLoc] + v;
          outF[((size_t)b * M + m) * HW + nLoc] = y;
          pk.u[r] = f2bf(y);
        }
        *(v4u*)(outH + ng * M + mRow) = pk.q;
      } else {
#pragma unroll
        for (int r = 0; r < 8; ++r) {
          int m = mRow + r;
          size_t oi = ((size_t)b * M + m) * HW + nLoc;
          outF[oi] = acc[mf][nf][r] + p0[m] + resid[oi];
        }
      }
    }
  }
}

// ---------------------------------------------------------------- depthwise 5x5 pad 2 (768 ch, f32, [b][c][n])
__global__ __launch_bounds__(256) void k_dw5(const float* __restrict__ in,
                                             const float* __restrict__ w,
                                             float* __restrict__ out) {
  int bc = blockIdx.x;                 // b*768 + c
  int c  = bc % 768;
  int n  = blockIdx.y * 256 + threadIdx.x;
  int y  = n >> 6, x = n & 63;
  const float* src = in + (size_t)bc * 4096;
  const float* wp  = w + c * 25;
  float acc = 0.f;
#pragma unroll
  for (int dy = -2; dy <= 2; ++dy) {
    int yy = y + dy;
    if (yy < 0 || yy > 63) continue;
#pragma unroll
    for (int dx = -2; dx <= 2; ++dx) {
      int xx = x + dx;
      if (xx < 0 || xx > 63) continue;
      acc += wp[(dy + 2) * 5 + (dx + 2)] * src[yy * 64 + xx];
    }
  }
  out[(size_t)bc * 4096 + n] = acc;
}

// ---------------------------------------------------------------- grouped 32x32 pointwise (24 groups, f32)
__global__ __launch_bounds__(256) void k_aggpw(const float* __restrict__ in,
                                               const float* __restrict__ w,
                                               float* __restrict__ out) {
  int bc = blockIdx.x;                 // b*768 + c_out
  int c  = bc % 768;
  int b  = bc / 768;
  int g  = c >> 5, o = c & 31;
  int n  = blockIdx.y * 256 + threadIdx.x;
  const float* src = in + ((size_t)b * 768 + g * 32) * 4096 + n;
  const float* wp  = w + (g * 32 + o) * 32;
  float acc = 0.f;
#pragma unroll
  for (int d = 0; d < 32; ++d) acc += wp[d] * src[(size_t)d * 4096];
  out[(size_t)bc * 4096 + n] = acc;
}

// ---------------------------------------------------------------- linear attention: kv = v1 @ k^T (33x32 per head)
__global__ __launch_bounds__(256) void k_attkv(const float* __restrict__ qkv,
                                               const float* __restrict__ agg,
                                               float* __restrict__ kv) {
  __shared__ float ks[32 * 128];
  __shared__ float vs[33 * 128];
  int bh = blockIdx.x;                 // b*16 + h
  int b  = bh >> 4, h = bh & 15;
  const float* src = (h < 8) ? qkv + ((size_t)b * 768 + h * 96) * 4096
                             : agg + ((size_t)b * 768 + (h - 8) * 96) * 4096;
  int tid = threadIdx.x;
  float acc[5] = {0.f, 0.f, 0.f, 0.f, 0.f};
  int n0 = blockIdx.y * 512;
  for (int ch = 0; ch < 4; ++ch) {
    int nb = n0 + ch * 128;
    for (int idx = tid; idx < 65 * 128; idx += 256) {
      int r = idx >> 7, col = idx & 127;
      if (r < 32)      ks[r * 128 + col] = fmaxf(src[(size_t)(32 + r) * 4096 + nb + col], 0.f);
      else if (r < 64) vs[(r - 32) * 128 + col] = src[(size_t)(64 + (r - 32)) * 4096 + nb + col];
      else             vs[32 * 128 + col] = 1.0f;
    }
    __syncthreads();
#pragma unroll
    for (int j = 0; j < 5; ++j) {
      int p = tid + j * 256;
      if (p < 1056) {
        int e = p >> 5, d = p & 31;
        const float* vr = &vs[e * 128];
        const float* kr = &ks[d * 128];
        float s = 0.f;
#pragma unroll 8
        for (int c2 = 0; c2 < 128; ++c2) s += vr[c2] * kr[c2];
        acc[j] += s;
      }
    }
    __syncthreads();
  }
#pragma unroll
  for (int j = 0; j < 5; ++j) {
    int p = tid + j * 256;
    if (p < 1056) atomicAdd(&kv[(size_t)bh * 1056 + p], acc[j]);
  }
}

// ---------------------------------------------------------------- attention apply -> attoutT bf16 [b*HW+n][512]
__global__ __launch_bounds__(256) void k_attapply(const float* __restrict__ qkv,
                                                  const float* __restrict__ agg,
                                                  const float* __restrict__ kv,
                                                  u16* __restrict__ outT) {
  __shared__ float kvs[1056];
  int bh = blockIdx.x;
  int b  = bh >> 4, h = bh & 15;
  const float* src = (h < 8) ? qkv + ((size_t)b * 768 + h * 96) * 4096
                             : agg + ((size_t)b * 768 + (h - 8) * 96) * 4096;
  int tid = threadIdx.x;
  for (int i = tid; i < 1056; i += 256) kvs[i] = kv[(size_t)bh * 1056 + i];
  __syncthreads();
  int n = blockIdx.y * 256 + tid;
  float acc[33];
#pragma unroll
  for (int e = 0; e < 33; ++e) acc[e] = 0.f;
#pragma unroll 4
  for (int d = 0; d < 32; ++d) {
    float q = fmaxf(src[(size_t)d * 4096 + n], 0.f);
#pragma unroll
    for (int e = 0; e < 33; ++e) acc[e] += kvs[e * 32 + d] * q;
  }
  float rden = 1.f / (acc[32] + 1e-15f);
  union { u16 u[32]; v4u q[4]; } pk;
#pragma unroll
  for (int e = 0; e < 32; ++e) pk.u[e] = f2bf(acc[e] * rden);
  v4u* dst = (v4u*)(outT + ((size_t)b * 4096 + n) * 512 + h * 32);
#pragma unroll
  for (int q2 = 0; q2 < 4; ++q2) dst[q2] = pk.q[q2];
}

// ---------------------------------------------------------------- depthwise 3x3 + GLU, [n][c] layouts (bf16)
// hT : [B*HW][2048] (a: c<1024, g: c>=1024),  gluT : [B*HW][1024]
__global__ __launch_bounds__(256) void k_dw3glu(const u16* __restrict__ hT,
                                                const float* __restrict__ w,
                                                const float* __restrict__ bdw,
                                                u16* __restrict__ gluT) {
  int bhw = blockIdx.x;                // b*4096 + n
  int n   = bhw & 4095;
  int y   = n >> 6, x = n & 63;
  int tid = threadIdx.x;
  float a[4], g[4];
#pragma unroll
  for (int i = 0; i < 4; ++i) {
    a[i] = bdw[tid + i * 256];
    g[i] = bdw[1024 + tid + i * 256];
  }
#pragma unroll
  for (int dy = -1; dy <= 1; ++dy) {
    int yy = y + dy;
    if (yy < 0 || yy > 63) continue;
#pragma unroll
    for (int dx = -1; dx <= 1; ++dx) {
      int xx = x + dx;
      if (xx < 0 || xx > 63) continue;
      const u16* row = hT + (size_t)(bhw + dy * 64 + dx) * 2048;
      int wi = (dy + 1) * 3 + (dx + 1);
#pragma unroll
      for (int i = 0; i < 4; ++i) {
        int c = tid + i * 256;
        a[i] += w[c * 9 + wi] * bf2f(row[c]);
        g[i] += w[(c + 1024) * 9 + wi] * bf2f(row[c + 1024]);
      }
    }
  }
  u16* orow = gluT + (size_t)bhw * 1024;
#pragma unroll
  for (int i = 0; i < 4; ++i) {
    int c = tid + i * 256;
    float o = a[i] * (g[i] / (1.f + __expf(-g[i])));    // a * silu(g)
    orow[c] = f2bf(o);
  }
}

// ---------------------------------------------------------------- launch
extern "C" void kernel_launch(void* const* d_in, const int* in_sizes, int n_in,
                              void* d_out, int out_size, void* d_ws, size_t ws_size,
                              hipStream_t stream) {
  (void)in_sizes; (void)n_in; (void)out_size; (void)ws_size;
  const float* x      = (const float*)d_in[0];
  const float* w_qkv  = (const float*)d_in[1];
  const float* w_aggd = (const float*)d_in[2];
  const float* w_aggp = (const float*)d_in[3];
  const float* w_proj = (const float*)d_in[4];
  const float* gamma  = (const float*)d_in[5];
  const float* beta   = (const float*)d_in[6];
  const float* mean   = (const float*)d_in[7];
  const float* var    = (const float*)d_in[8];
  const float* w_inv  = (const float*)d_in[9];
  const float* b_inv  = (const float*)d_in[10];
  const float* w_dw   = (const float*)d_in[11];
  const float* b_dw   = (const float*)d_in[12];
  const float* w_pt   = (const float*)d_in[13];
  const float* b_pt   = (const float*)d_in[14];
  float* out = (float*)d_out;

  const int Bb = 8, C = 256, HW = 4096;
  const int Nn = Bb * HW;              // 32768

  char* ws = (char*)d_ws;
  size_t off = 0;
  auto take = [&](size_t bytes) -> char* {
    char* p = ws + off;
    off = (off + bytes + 255) & ~(size_t)255;
    return p;
  };

  u16*   xT     = (u16*)  take((size_t)Nn * C * 2);          // [n][256]
  u16*   wqkvb  = (u16*)  take((size_t)768 * 256 * 2);
  u16*   wprojb = (u16*)  take((size_t)256 * 512 * 2);
  u16*   winvb  = (u16*)  take((size_t)2048 * 256 * 2);
  u16*   wptb   = (u16*)  take((size_t)256 * 1024 * 2);
  float* qkv    = (float*)take((size_t)Bb * 768 * HW * 4);   // [b][c][n]
  float* aggdw  = (float*)take((size_t)Bb * 768 * HW * 4);
  float* aggpw  = (float*)take((size_t)Bb * 768 * HW * 4);
  float* kvb    = (float*)take((size_t)64 * 1056 * 4);
  u16*   attT   = (u16*)  take((size_t)Nn * 512 * 2);        // [n][512]
  float* y1     = (float*)take((size_t)Bb * 256 * HW * 4);   // [b][m][n]
  u16*   y1T    = (u16*)  take((size_t)Nn * 256 * 2);        // [n][256]
  u16*   hT     = (u16*)  take((size_t)Nn * 2048 * 2);       // [n][2048]
  u16*   gluT   = (u16*)  take((size_t)Nn * 1024 * 2);       // [n][1024]

  // activation transpose-convert + weight converts
  k_cvtT<<<dim3(HW / 64, C / 64, Bb), 256, 0, stream>>>(x, xT);
  k_cvt<<<dim3((768 * 256 + 255) / 256), 256, 0, stream>>>(w_qkv, wqkvb, 768 * 256);
  k_cvt<<<dim3((256 * 512 + 255) / 256), 256, 0, stream>>>(w_proj, wprojb, 256 * 512);
  k_cvt<<<dim3((2048 * 256 + 255) / 256), 256, 0, stream>>>(w_inv, winvb, 2048 * 256);
  k_cvt<<<dim3((256 * 1024 + 255) / 256), 256, 0, stream>>>(w_pt, wptb, 256 * 1024);

  // qkv = w_qkv @ x                                (M=768, K=256)
  k_gemm<0><<<dim3(Nn / 128, 768 / 128), 256, 0, stream>>>(
      wqkvb, xT, 768, 256, HW, qkv, nullptr,
      nullptr, nullptr, nullptr, nullptr, nullptr);

  // depthwise 5x5 on qkv
  k_dw5<<<dim3(Bb * 768, 16), 256, 0, stream>>>(qkv, w_aggd, aggdw);

  // grouped 32x32 pointwise
  k_aggpw<<<dim3(Bb * 768, 16), 256, 0, stream>>>(aggdw, w_aggp, aggpw);

  // linear attention
  k_zero<<<dim3((64 * 1056 + 255) / 256), 256, 0, stream>>>(kvb, 64 * 1056);
  k_attkv<<<dim3(64, 8), 256, 0, stream>>>(qkv, aggpw, kvb);
  k_attapply<<<dim3(64, 16), 256, 0, stream>>>(qkv, aggpw, kvb, attT);

  // proj + BN + residual -> y1 (f32 [b][m][n]) and y1T (bf16 [n][m])   (M=256, K=512)
  k_gemm<2><<<dim3(Nn / 128, 256 / 128), 256, 0, stream>>>(
      wprojb, attT, 256, 512, HW, y1, y1T,
      gamma, beta, mean, var, x);

  // inverted bottleneck: w_inv @ y1 + b_inv, SiLU -> hT bf16 [n][2048]   (M=2048, K=256)
  k_gemm<1><<<dim3(Nn / 128, 2048 / 128), 256, 0, stream>>>(
      winvb, y1T, 2048, 256, HW, nullptr, hT,
      b_inv, nullptr, nullptr, nullptr, nullptr);

  // depthwise 3x3 + bias + GLU -> gluT bf16 [n][1024]
  k_dw3glu<<<dim3(Bb * HW), 256, 0, stream>>>(hT, w_dw, b_dw, gluT);

  // pointwise: w_pt @ glu + b_pt + y1 -> out       (M=256, K=1024)
  k_gemm<3><<<dim3(Nn / 128, 256 / 128), 256, 0, stream>>>(
      wptb, gluT, 256, 1024, HW, out, nullptr,
      b_pt, nullptr, nullptr, nullptr, y1);
}
